// GPRGNN_3461743641084
// MI455X (gfx1250) — matrix operand order repaired and checked by reading, not compiled, not run on hardware
//
#include <hip/hip_runtime.h>
#include <cstddef>

typedef __bf16 bf16_t;
typedef bf16_t v16bf __attribute__((ext_vector_type(16)));
typedef float  v8f   __attribute__((ext_vector_type(8)));

#define WMMA_BF16(acc, a, b) \
  (acc) = __builtin_amdgcn_wmma_f32_16x16x32_bf16(false, (a), false, (b), (short)0, (acc), false, false)

// ---------------- graph normalization ----------------
__global__ void k_zero_u32(unsigned* __restrict__ p, int n) {
  int t = blockIdx.x * blockDim.x + threadIdx.x;
  if (t < n) p[t] = 0u;
}

__global__ void k_count_deg(const long long* __restrict__ ei, unsigned* __restrict__ cnt, int E) {
  int t = blockIdx.x * blockDim.x + threadIdx.x;
  if (t < E) atomicAdd(&cnt[(int)ei[(size_t)E + t]], 1u);
}

__global__ void k_dinv(const unsigned* __restrict__ cnt, float* __restrict__ dinv, int n) {
  int t = blockIdx.x * blockDim.x + threadIdx.x;
  if (t < n) dinv[t] = rsqrtf((float)(cnt[t] + 1u));  // +1 = self loop
}

__global__ void k_norm(const long long* __restrict__ ei, const float* __restrict__ dinv,
                       float* __restrict__ nrm, int E) {
  int t = blockIdx.x * blockDim.x + threadIdx.x;
  if (t < E) nrm[t] = dinv[(int)ei[t]] * dinv[(int)ei[(size_t)E + t]];
}

// ---------------- weight packing into WMMA B-fragment order ----------------
// layout: [kt][nt][lane][j], j=0..15 contiguous bf16 per lane (32B loads)
// lane: n_local = lane&15, half = lane>>4 ; k = kt*32 + half*16 + j ; n = nt*16 + n_local
__global__ void k_pack_weights(const float* __restrict__ W1, const float* __restrict__ W2,
                               bf16_t* __restrict__ W1p, bf16_t* __restrict__ W2p) {
  const int TOT1 = 16 * 4 * 32 * 16;  // 512x64
  const int TOT2 = 2 * 4 * 32 * 16;   // 64x64
  for (int idx = blockIdx.x * blockDim.x + threadIdx.x; idx < TOT1 + TOT2;
       idx += gridDim.x * blockDim.x) {
    if (idx < TOT1) {
      int j = idx & 15, lane = (idx >> 4) & 31, nt = (idx >> 9) & 3, kt = idx >> 11;
      int k = kt * 32 + (lane >> 4) * 16 + j;
      int n = nt * 16 + (lane & 15);
      W1p[idx] = (bf16_t)W1[k * 64 + n];
    } else {
      int i2 = idx - TOT1;
      int j = i2 & 15, lane = (i2 >> 4) & 31, nt = (i2 >> 9) & 3, kt = i2 >> 11;
      int k = kt * 32 + (lane >> 4) * 16 + j;
      int n = nt * 16 + (lane & 15);
      W2p[i2] = (bf16_t)W2[k * 64 + n];
    }
  }
}

// ---------------- fused MLP: h = relu(X@W1+b1)@W2+b2 ; hidden = temp[0]*h ----
// one wave32 -> one 16-row x 64-col output tile; 8 waves / block
__global__ __launch_bounds__(256) void k_mlp_wmma(
    const float* __restrict__ X, const bf16_t* __restrict__ W1p, const float* __restrict__ b1,
    const bf16_t* __restrict__ W2p, const float* __restrict__ b2, const float* __restrict__ temp,
    float* __restrict__ h_out, float* __restrict__ hidden_out, int nTiles, int N) {
  __shared__ bf16_t lds[8][16][64];

  const int lane = threadIdx.x & 31;
  const int wave = threadIdx.x >> 5;
  int tile = blockIdx.x * 8 + wave;
  const bool valid = tile < nTiles;
  if (!valid) tile = nTiles - 1;  // clamp: compute on last tile, skip stores
  const int row0 = tile * 16;
  const int m = lane & 15;     // A row / C column-in-tile
  const int half = lane >> 4;  // K half selector

  v8f acc0 = {}, acc1 = {}, acc2 = {}, acc3 = {};

  // ---- GEMM1: X[16x512] @ W1[512x64] ----
  const float* xrow = X + (size_t)(row0 + m) * 512;
  for (int kb = 0; kb < 512; kb += 32) {
    v16bf va;
#pragma unroll
    for (int j = 0; j < 8; ++j) {
      va[j]     = (bf16_t)xrow[kb + half * 16 + j];
      va[8 + j] = (bf16_t)xrow[kb + half * 16 + 8 + j];
    }
    const bf16_t* bb = W1p + (size_t)(kb >> 5) * 2048 + lane * 16;
    v16bf vb0 = *(const v16bf*)(bb + 0 * 512);
    v16bf vb1 = *(const v16bf*)(bb + 1 * 512);
    v16bf vb2 = *(const v16bf*)(bb + 2 * 512);
    v16bf vb3 = *(const v16bf*)(bb + 3 * 512);
    WMMA_BF16(acc0, va, vb0);
    WMMA_BF16(acc1, va, vb1);
    WMMA_BF16(acc2, va, vb2);
    WMMA_BF16(acc3, va, vb3);
  }

  // bias + relu, reshape C-layout -> row-major bf16 tile in LDS
#pragma unroll
  for (int r = 0; r < 8; ++r) {
    const int mm = r + 8 * half;
    float v0 = fmaxf(acc0[r] + b1[0 * 16 + m], 0.f);
    float v1 = fmaxf(acc1[r] + b1[1 * 16 + m], 0.f);
    float v2 = fmaxf(acc2[r] + b1[2 * 16 + m], 0.f);
    float v3 = fmaxf(acc3[r] + b1[3 * 16 + m], 0.f);
    lds[wave][mm][0 * 16 + m] = (bf16_t)v0;
    lds[wave][mm][1 * 16 + m] = (bf16_t)v1;
    lds[wave][mm][2 * 16 + m] = (bf16_t)v2;
    lds[wave][mm][3 * 16 + m] = (bf16_t)v3;
  }
  __syncthreads();

  // ---- GEMM2: h1[16x64] @ W2[64x64] ----
  v8f d0 = {}, d1 = {}, d2 = {}, d3 = {};
#pragma unroll
  for (int kk = 0; kk < 2; ++kk) {
    const int kb2 = kk * 32;
    v16bf va2;
#pragma unroll
    for (int j = 0; j < 8; ++j) {
      va2[j]     = lds[wave][m][kb2 + half * 16 + j];
      va2[8 + j] = lds[wave][m][kb2 + half * 16 + 8 + j];
    }
    const bf16_t* bb = W2p + (size_t)kk * 2048 + lane * 16;
    v16bf vb0 = *(const v16bf*)(bb + 0 * 512);
    v16bf vb1 = *(const v16bf*)(bb + 1 * 512);
    v16bf vb2 = *(const v16bf*)(bb + 2 * 512);
    v16bf vb3 = *(const v16bf*)(bb + 3 * 512);
    WMMA_BF16(d0, va2, vb0);
    WMMA_BF16(d1, va2, vb1);
    WMMA_BF16(d2, va2, vb2);
    WMMA_BF16(d3, va2, vb3);
  }

  if (valid) {
    const float t0 = temp[0];
#pragma unroll
    for (int r = 0; r < 8; ++r) {
      const int mm = r + 8 * half;
      const int row = row0 + mm;
      if (row < N) {
        float v0 = d0[r] + b2[0 * 16 + m];
        float v1 = d1[r] + b2[1 * 16 + m];
        float v2 = d2[r] + b2[2 * 16 + m];
        float v3 = d3[r] + b2[3 * 16 + m];
        float* hp = h_out + (size_t)row * 64;
        float* gp = hidden_out + (size_t)row * 64;
        hp[0 * 16 + m] = v0; gp[0 * 16 + m] = t0 * v0;
        hp[1 * 16 + m] = v1; gp[1 * 16 + m] = t0 * v1;
        hp[2 * 16 + m] = v2; gp[2 * 16 + m] = t0 * v2;
        hp[3 * 16 + m] = v3; gp[3 * 16 + m] = t0 * v3;
      }
    }
  }
}

// ---------------- propagation ----------------
// dst[i,:] = dinv[i]^2 * src[i,:]   (self-loop term; also fully initializes dst)
__global__ void k_selfloop_init(float* __restrict__ dst, const float* __restrict__ src,
                                const float* __restrict__ dinv, int total) {
  int t = blockIdx.x * blockDim.x + threadIdx.x;
  if (t >= total) return;
  float di = dinv[t >> 6];
  dst[t] = di * di * src[t];
}

// dst[col,:] += norm[e] * src[row,:]  (64 channels per edge; wave32-coalesced)
__global__ __launch_bounds__(256) void k_edge_scatter(
    const long long* __restrict__ ei, const float* __restrict__ nrm,
    const float* __restrict__ src, float* __restrict__ dst, int E) {
  int t = blockIdx.x * blockDim.x + threadIdx.x;
  int e = t >> 6;
  if (e >= E) return;
  int ch = t & 63;
  int r = (int)ei[e];
  int c = (int)ei[(size_t)E + e];
  float w = nrm[e];
  atomicAdd(&dst[(size_t)c * 64 + ch], w * src[(size_t)r * 64 + ch]);
}

__global__ void k_axpy(float* __restrict__ out, const float* __restrict__ buf,
                       const float* __restrict__ temp, int tidx, int total) {
  int t = blockIdx.x * blockDim.x + threadIdx.x;
  if (t < total) out[t] += temp[tidx] * buf[t];
}

// ---------------- log_softmax over 64 classes, one wave32 per row ----------------
__global__ __launch_bounds__(256) void k_log_softmax(float* __restrict__ out, int nRows) {
  int lane = threadIdx.x & 31;
  int row = blockIdx.x * 8 + (threadIdx.x >> 5);
  if (row >= nRows) return;
  float* p = out + (size_t)row * 64;
  float a = p[lane];
  float b = p[lane + 32];
  float mx = fmaxf(a, b);
#pragma unroll
  for (int o = 16; o > 0; o >>= 1) mx = fmaxf(mx, __shfl_xor(mx, o, 32));
  float s = expf(a - mx) + expf(b - mx);
#pragma unroll
  for (int o = 16; o > 0; o >>= 1) s += __shfl_xor(s, o, 32);
  float l = mx + logf(s);
  p[lane] = a - l;
  p[lane + 32] = b - l;
}

// ---------------- driver ----------------
extern "C" void kernel_launch(void* const* d_in, const int* in_sizes, int n_in,
                              void* d_out, int out_size, void* d_ws, size_t ws_size,
                              hipStream_t stream) {
  const float*      X    = (const float*)d_in[0];
  const long long*  ei   = (const long long*)d_in[1];
  const float*      W1   = (const float*)d_in[2];
  const float*      b1   = (const float*)d_in[3];
  const float*      W2   = (const float*)d_in[4];
  const float*      b2   = (const float*)d_in[5];
  const float*      temp = (const float*)d_in[6];
  float*            out  = (float*)d_out;

  const int N = in_sizes[0] / 512;
  const int E = in_sizes[1] / 2;
  const int K = in_sizes[6] - 1;

  // workspace carve-out (256B aligned)
  size_t off = 0;
  auto carve = [&](size_t bytes) {
    void* p = (char*)d_ws + off;
    off += (bytes + 255) & ~(size_t)255;
    return p;
  };
  unsigned* degCnt = (unsigned*)carve((size_t)N * 4);
  float*    dinv   = (float*)carve((size_t)N * 4);
  float*    nrm    = (float*)carve((size_t)E * 4);
  bf16_t*   W1p    = (bf16_t*)carve((size_t)32768 * 2);
  bf16_t*   W2p    = (bf16_t*)carve((size_t)4096 * 2);
  float*    bufA   = (float*)carve((size_t)N * 64 * 4);
  float*    bufB   = (float*)carve((size_t)N * 64 * 4);

  const int total = N * 64;

  // graph norm
  k_zero_u32<<<(N + 255) / 256, 256, 0, stream>>>(degCnt, N);
  k_count_deg<<<(E + 255) / 256, 256, 0, stream>>>(ei, degCnt, E);
  k_dinv<<<(N + 255) / 256, 256, 0, stream>>>(degCnt, dinv, N);
  k_norm<<<(E + 255) / 256, 256, 0, stream>>>(ei, dinv, nrm, E);

  // MLP with WMMA
  k_pack_weights<<<144, 256, 0, stream>>>(W1, W2, W1p, W2p);
  const int nTiles = (N + 15) / 16;
  k_mlp_wmma<<<(nTiles + 7) / 8, 256, 0, stream>>>(X, W1p, b1, W2p, b2, temp,
                                                   bufA, out, nTiles, N);

  // K hops of GCN-normalized propagation with PPR accumulation
  float* cur = bufA;
  float* nxt = bufB;
  const unsigned gridElem = (unsigned)((total + 255) / 256);
  const unsigned gridEdge = (unsigned)(((size_t)E * 64 + 255) / 256);
  for (int k = 0; k < K; ++k) {
    k_selfloop_init<<<gridElem, 256, 0, stream>>>(nxt, cur, dinv, total);
    k_edge_scatter<<<gridEdge, 256, 0, stream>>>(ei, nrm, cur, nxt, E);
    k_axpy<<<gridElem, 256, 0, stream>>>(out, nxt, temp, k + 1, total);
    float* t = cur; cur = nxt; nxt = t;
  }

  k_log_softmax<<<(unsigned)((N + 7) / 8), 256, 0, stream>>>(out, N);
}
